// _NonLocalBlockND_54168127537869
// MI455X (gfx1250) — compile-verified, matrix operand order checked by reading, not run on hardware
//
#include <hip/hip_runtime.h>
#include <hip/hip_bf16.h>
#include <math.h>

// ---------------------------------------------------------------------------
// NonLocal block, algebraically refactored (linear "attention" => associative):
//   T,P,G = proj(x)                    (f16 WMMA GEMM, K=512)
//   M_b   = P_b @ G_b^T / N            (256x256 per batch, K=2304)
//   Weff  = w_w @ M_b^T                (512x256 per batch, K=256)
//   out   = BN(Weff @ T_b + w_b) + x   (512x2304 per batch, K=256)
// All GEMMs: f16 inputs, f32 accumulation via v_wmma_f32_16x16x32_f16.
// K-loop is ping-pong unrolled by 2: loads of one buffer overlap WMMAs of the
// other, with no register-copy rotation (requires K/32 even, K>=64 -- true
// for all GEMMs here: K in {512, 2304, 256}).
// ---------------------------------------------------------------------------

typedef __attribute__((ext_vector_type(16))) _Float16 v16h;
typedef __attribute__((ext_vector_type(8)))  _Float16 v8h;
typedef __attribute__((ext_vector_type(8)))  float    v8f;

#define NB 16
#define CC 512
#define CI 256
#define NN 2304

// ---- fragment loader -------------------------------------------------------
// Loads a 16x32 f16 WMMA operand fragment. Memory layout: element (r, k) at
// base[r*ld + k] (K contiguous). Matches CDNA5 16-bit A/B layout:
//   row/col = lane % 16 ; half h -> k = 16*(h/8) + 8*(lane/16) + (h%8)
// i.e. two contiguous 16-byte chunks per lane.
__device__ __forceinline__ v16h load_fragK(const _Float16* __restrict__ base,
                                           int ld, int rc, int k0, int lane) {
  const int r = rc + (lane & 15);
  const int g = (lane >> 4) & 1;
  const _Float16* p = base + (size_t)r * (size_t)ld + (size_t)(k0 + 8 * g);
  v8h lo = *(const v8h*)(p);        // k = k0 + 8g      .. +8
  v8h hi = *(const v8h*)(p + 16);   // k = k0 + 16 + 8g .. +8
  return __builtin_shufflevector(lo, hi, 0,1,2,3,4,5,6,7,8,9,10,11,12,13,14,15);
}

// ---- core GEMM accumulate loop (ping-pong double buffered) -----------------
// A: row-major [M,K] (lda=K-stride). Bm: "n-major" [N,K]: B[k,n]=Bm[n*ldb+k].
// Requires: K % 64 == 0 (even number of 32-wide K-steps), K >= 64.
template<int MT, int NT>
__device__ __forceinline__ void gemm_accum(const _Float16* __restrict__ A, int lda,
                                           const _Float16* __restrict__ Bm, int ldb,
                                           int m0, int n0, int K, int lane,
                                           v8f acc[MT][NT]) {
  v16h a0[MT], b0[NT], a1[MT], b1[NT];

#define LOAD_BUF(abuf, bbuf, kk)                                               \
  {                                                                            \
    _Pragma("unroll")                                                          \
    for (int mi = 0; mi < MT; ++mi) abuf[mi] = load_fragK(A, lda, m0 + 16*mi, (kk), lane); \
    _Pragma("unroll")                                                          \
    for (int ni = 0; ni < NT; ++ni) bbuf[ni] = load_fragK(Bm, ldb, n0 + 16*ni, (kk), lane); \
  }
#define WMMA_BUF(abuf, bbuf)                                                   \
  {                                                                            \
    _Pragma("unroll")                                                          \
    for (int mi = 0; mi < MT; ++mi)                                            \
      _Pragma("unroll")                                                        \
      for (int ni = 0; ni < NT; ++ni)                                          \
        acc[mi][ni] = __builtin_amdgcn_wmma_f32_16x16x32_f16(                  \
            false, abuf[mi], false, bbuf[ni], (short)0, acc[mi][ni], false, false); \
  }

  LOAD_BUF(a0, b0, 0)
  int k0 = 32;
  for (; k0 < K - 32; k0 += 64) {
    LOAD_BUF(a1, b1, k0)        // loads overlap WMMAs on buf0
    WMMA_BUF(a0, b0)
    LOAD_BUF(a0, b0, k0 + 32)   // loads overlap WMMAs on buf1
    WMMA_BUF(a1, b1)
  }
  LOAD_BUF(a1, b1, K - 32)
  WMMA_BUF(a0, b0)
  WMMA_BUF(a1, b1)

#undef LOAD_BUF
#undef WMMA_BUF
}

// ---- kernel: convert weights to f16 ---------------------------------------
// Wstack rows: [0,256)=theta_w, [256,512)=phi_w, [512,768)=g_w  (each [Ci,C])
__global__ void k_wconv(const float* __restrict__ theta_w, const float* __restrict__ phi_w,
                        const float* __restrict__ g_w, const float* __restrict__ w_w,
                        _Float16* __restrict__ Wstack, _Float16* __restrict__ Ww16) {
  int i = blockIdx.x * 256 + threadIdx.x;
  const int NW = 768 * CC;
  if (i < NW) {
    int o = i >> 9, c = i & 511;
    float v = (o < 256) ? theta_w[(size_t)o * CC + c]
            : (o < 512) ? phi_w[(size_t)(o - 256) * CC + c]
                        : g_w[(size_t)(o - 512) * CC + c];
    Wstack[i] = (_Float16)v;
  } else {
    int j = i - NW;                      // < 512*256
    Ww16[j] = (_Float16)w_w[j];
  }
}

// ---- kernel: transpose+convert x -> Xt f16 [B, N, C] (K=C contiguous) -----
__global__ void k_transpose(const float* __restrict__ x, _Float16* __restrict__ Xt) {
  __shared__ float tile[32][33];
  const int b  = blockIdx.z;
  const int c0 = blockIdx.x * 32;
  const int n0 = blockIdx.y * 32;
  const int tx = threadIdx.x, ty = threadIdx.y;
  const float* src = x + ((size_t)b * CC + c0) * NN + n0;
#pragma unroll
  for (int r = 0; r < 4; ++r)
    tile[ty + 8*r][tx] = src[(size_t)(ty + 8*r) * NN + tx];   // coalesced in n
  __syncthreads();
  _Float16* dst = Xt + ((size_t)b * NN + n0) * CC + c0;
#pragma unroll
  for (int r = 0; r < 4; ++r)
    dst[(size_t)(ty + 8*r) * CC + tx] = (_Float16)tile[tx][ty + 8*r]; // coalesced in c
}

// ---- kernel: fused 3-way projection (theta/phi/g), bias included ----------
// WG = 128 thr = 4 waves (2x2); wave tile 64x32; WG tile 128x64.
// Outputs: Tt [B,N,Ci] (theta, transposed), P,G [B,Ci,N].
__global__ __launch_bounds__(128) void k_proj(const _Float16* __restrict__ Wstack,
                                              const _Float16* __restrict__ Xt,
                                              const float* __restrict__ theta_b,
                                              const float* __restrict__ phi_b,
                                              const float* __restrict__ g_b,
                                              _Float16* __restrict__ Tt,
                                              _Float16* __restrict__ P,
                                              _Float16* __restrict__ G) {
  const int lane = threadIdx.x & 31, wave = threadIdx.x >> 5;
  const int wm = wave >> 1, wn = wave & 1;
  const int b  = blockIdx.z;
  const int m0 = blockIdx.x * 128 + wm * 64;     // o in [0,768)
  const int n0 = blockIdx.y * 64  + wn * 32;     // n in [0,2304)
  const _Float16* Xb = Xt + (size_t)b * NN * CC;

  v8f acc[4][2];
#pragma unroll
  for (int i = 0; i < 4; ++i)
#pragma unroll
    for (int j = 0; j < 2; ++j) { v8f z = {}; acc[i][j] = z; }

  gemm_accum<4,2>(Wstack, CC, Xb, CC, m0, n0, CC, lane, acc);

  const int g4 = (lane >> 4) & 1, nl = lane & 15;
#pragma unroll
  for (int mi = 0; mi < 4; ++mi)
#pragma unroll
    for (int ni = 0; ni < 2; ++ni)
#pragma unroll
      for (int v = 0; v < 8; ++v) {
        int o = m0 + 16*mi + v + 8*g4;
        int n = n0 + 16*ni + nl;
        float bias = (o < 256) ? theta_b[o] : (o < 512) ? phi_b[o - 256] : g_b[o - 512];
        float val = acc[mi][ni][v] + bias;
        if (o < 256)      Tt[((size_t)b * NN + n) * CI + o]            = (_Float16)val;
        else if (o < 512) P [((size_t)b * CI + (o - 256)) * NN + n]    = (_Float16)val;
        else              G [((size_t)b * CI + (o - 512)) * NN + n]    = (_Float16)val;
      }
}

// ---- kernel: M_b = P_b @ G_b^T / N  (256x256, K=2304) ----------------------
// wave tile 32x32; WG tile 64x64.
__global__ __launch_bounds__(128) void k_mgemm(const _Float16* __restrict__ P,
                                               const _Float16* __restrict__ G,
                                               _Float16* __restrict__ Mb) {
  const int lane = threadIdx.x & 31, wave = threadIdx.x >> 5;
  const int wm = wave >> 1, wn = wave & 1;
  const int b  = blockIdx.z;
  const int m0 = blockIdx.x * 64 + wm * 32;      // i
  const int n0 = blockIdx.y * 64 + wn * 32;      // j
  const _Float16* Pb = P + (size_t)b * CI * NN;
  const _Float16* Gb = G + (size_t)b * CI * NN;

  v8f acc[2][2];
#pragma unroll
  for (int i = 0; i < 2; ++i)
#pragma unroll
    for (int j = 0; j < 2; ++j) { v8f z = {}; acc[i][j] = z; }

  gemm_accum<2,2>(Pb, NN, Gb, NN, m0, n0, NN, lane, acc);

  const float inv = 1.0f / (float)NN;
  const int g4 = (lane >> 4) & 1, nl = lane & 15;
#pragma unroll
  for (int mi = 0; mi < 2; ++mi)
#pragma unroll
    for (int ni = 0; ni < 2; ++ni)
#pragma unroll
      for (int v = 0; v < 8; ++v) {
        int i = m0 + 16*mi + v + 8*g4;
        int j = n0 + 16*ni + nl;
        Mb[(size_t)b * CI * CI + (size_t)i * CI + j] = (_Float16)(acc[mi][ni][v] * inv);
      }
}

// ---- kernel: Weff_b = w_w @ M_b^T  (512x256, K=256) ------------------------
__global__ __launch_bounds__(128) void k_weff(const _Float16* __restrict__ Ww16,
                                              const _Float16* __restrict__ Mb,
                                              _Float16* __restrict__ Weff) {
  const int lane = threadIdx.x & 31, wave = threadIdx.x >> 5;
  const int wm = wave >> 1, wn = wave & 1;
  const int b  = blockIdx.z;
  const int m0 = blockIdx.x * 64 + wm * 32;      // c
  const int n0 = blockIdx.y * 64 + wn * 32;      // ci
  const _Float16* Mbb = Mb + (size_t)b * CI * CI;  // B[k=j, n=ci] = M[ci, j]

  v8f acc[2][2];
#pragma unroll
  for (int i = 0; i < 2; ++i)
#pragma unroll
    for (int j = 0; j < 2; ++j) { v8f z = {}; acc[i][j] = z; }

  gemm_accum<2,2>(Ww16, CI, Mbb, CI, m0, n0, CI, lane, acc);

  const int g4 = (lane >> 4) & 1, nl = lane & 15;
#pragma unroll
  for (int mi = 0; mi < 2; ++mi)
#pragma unroll
    for (int ni = 0; ni < 2; ++ni)
#pragma unroll
      for (int v = 0; v < 8; ++v) {
        int c  = m0 + 16*mi + v + 8*g4;
        int ci = n0 + 16*ni + nl;
        Weff[(size_t)b * CC * CI + (size_t)c * CI + ci] = (_Float16)acc[mi][ni][v];
      }
}

// ---- kernel: out = BN(Weff_b @ T_b + w_b) + x  (512x2304, K=256) ----------
__global__ __launch_bounds__(128) void k_final(const _Float16* __restrict__ Weff,
                                               const _Float16* __restrict__ Tt,
                                               const float* __restrict__ w_b,
                                               const float* __restrict__ bn_gamma,
                                               const float* __restrict__ bn_beta,
                                               const float* __restrict__ bn_mean,
                                               const float* __restrict__ bn_var,
                                               const float* __restrict__ x,
                                               float* __restrict__ out) {
  const int lane = threadIdx.x & 31, wave = threadIdx.x >> 5;
  const int wm = wave >> 1, wn = wave & 1;
  const int b  = blockIdx.z;
  const int m0 = blockIdx.x * 128 + wm * 64;     // c
  const int n0 = blockIdx.y * 64  + wn * 32;     // n
  const _Float16* Ab = Weff + (size_t)b * CC * CI;       // [C, Ci] row-major
  const _Float16* Bb = Tt   + (size_t)b * NN * CI;       // B[k=ci, n] = Tt[n, ci]

  v8f acc[4][2];
#pragma unroll
  for (int i = 0; i < 4; ++i)
#pragma unroll
    for (int j = 0; j < 2; ++j) { v8f z = {}; acc[i][j] = z; }

  gemm_accum<4,2>(Ab, CI, Bb, CI, m0, n0, CI, lane, acc);

  const int g4 = (lane >> 4) & 1, nl = lane & 15;
#pragma unroll
  for (int mi = 0; mi < 4; ++mi)
#pragma unroll
    for (int ni = 0; ni < 2; ++ni)
#pragma unroll
      for (int v = 0; v < 8; ++v) {
        int c = m0 + 16*mi + v + 8*g4;
        int n = n0 + 16*ni + nl;
        float sc = bn_gamma[c] * rsqrtf(bn_var[c] + 1e-5f);
        float sh = bn_beta[c] - bn_mean[c] * sc;
        size_t idx = ((size_t)b * CC + c) * NN + n;
        out[idx] = (acc[mi][ni][v] + w_b[c]) * sc + sh + x[idx];
      }
}

// ---------------------------------------------------------------------------
extern "C" void kernel_launch(void* const* d_in, const int* in_sizes, int n_in,
                              void* d_out, int out_size, void* d_ws, size_t ws_size,
                              hipStream_t stream) {
  const float* x        = (const float*)d_in[0];
  const float* g_w      = (const float*)d_in[1];
  const float* g_b      = (const float*)d_in[2];
  const float* theta_w  = (const float*)d_in[3];
  const float* theta_b  = (const float*)d_in[4];
  const float* phi_w    = (const float*)d_in[5];
  const float* phi_b    = (const float*)d_in[6];
  const float* w_w      = (const float*)d_in[7];
  const float* w_b      = (const float*)d_in[8];
  const float* bn_gamma = (const float*)d_in[9];
  const float* bn_beta  = (const float*)d_in[10];
  const float* bn_mean  = (const float*)d_in[11];
  const float* bn_var   = (const float*)d_in[12];
  float* out = (float*)d_out;

  // workspace layout (bytes); all chunks 256B-aligned
  char* ws = (char*)d_ws;
  const size_t XT_OFF = 0;                                // f16 [B,N,C]   37,748,736
  const size_t WS_OFF = XT_OFF + (size_t)NB*NN*CC*2;      // f16 [768,512]    786,432
  const size_t WW_OFF = WS_OFF + (size_t)768*CC*2;        // f16 [512,256]    262,144
  const size_t TT_OFF = WW_OFF + (size_t)CC*CI*2;         // f16 [B,N,Ci]  18,874,368
  const size_t P_OFF  = TT_OFF + (size_t)NB*NN*CI*2;      // f16 [B,Ci,N]  18,874,368
  const size_t G_OFF  = P_OFF  + (size_t)NB*CI*NN*2;      // f16 [B,Ci,N]  18,874,368
  const size_t MB_OFF = G_OFF  + (size_t)NB*CI*NN*2;      // f16 [B,Ci,Ci]  2,097,152
  const size_t WE_OFF = MB_OFF + (size_t)NB*CI*CI*2;      // f16 [B,C,Ci]   4,194,304

  _Float16* Xt     = (_Float16*)(ws + XT_OFF);
  _Float16* Wstack = (_Float16*)(ws + WS_OFF);
  _Float16* Ww16   = (_Float16*)(ws + WW_OFF);
  _Float16* Tt     = (_Float16*)(ws + TT_OFF);
  _Float16* P      = (_Float16*)(ws + P_OFF);
  _Float16* G      = (_Float16*)(ws + G_OFF);
  _Float16* Mb     = (_Float16*)(ws + MB_OFF);
  _Float16* Weff   = (_Float16*)(ws + WE_OFF);

  // 1) convert weights (768*512 + 512*256 = 524288 elements)
  k_wconv<<<dim3(2048), dim3(256), 0, stream>>>(theta_w, phi_w, g_w, w_w, Wstack, Ww16);
  // 2) transpose+convert x -> [B, N, C] f16
  k_transpose<<<dim3(CC/32, NN/32, NB), dim3(32, 8), 0, stream>>>(x, Xt);
  // 3) fused projections: [768,512] @ [512,2304] per batch
  k_proj<<<dim3(768/128, NN/64, NB), dim3(128), 0, stream>>>(
      Wstack, Xt, theta_b, phi_b, g_b, Tt, P, G);
  // 4) M_b = P_b @ G_b^T / N
  k_mgemm<<<dim3(CI/64, CI/64, NB), dim3(128), 0, stream>>>(P, G, Mb);
  // 5) Weff_b = w_w @ M_b^T
  k_weff<<<dim3(CC/64, CI/64, NB), dim3(128), 0, stream>>>(Ww16, Mb, Weff);
  // 6) out = BN(Weff_b @ T_b + w_b) + x
  k_final<<<dim3(CC/128, NN/64, NB), dim3(128), 0, stream>>>(
      Weff, Tt, w_b, bn_gamma, bn_beta, bn_mean, bn_var, x, out);

  (void)in_sizes; (void)n_in; (void)out_size; (void)ws_size;
}
